// MobileBertSelfAttention_30812095381709
// MI455X (gfx1250) — matrix operand order repaired and checked by reading, not compiled, not run on hardware
//
#include <hip/hip_runtime.h>
#include <hip/hip_bf16.h>

// ---------------------------------------------------------------------------
// MobileBERT self-attention on gfx1250 (MI455X), bf16 WMMA + f32 accumulate.
// B=8, S=2048, H_MODEL=512, NUM_HEADS=8, HEAD_DIM=64.
// ---------------------------------------------------------------------------

typedef __attribute__((ext_vector_type(16))) __bf16 v16bf;
typedef __attribute__((ext_vector_type(8)))  float  v8f;

// gfx1250 async LDS copy path (ASYNCcnt-tracked). Builtin signature (from the
// compiler diagnostic): (v4i addrspace(1)*, v4i addrspace(3)*, imm offset, imm cpol)
#if __has_builtin(__builtin_amdgcn_global_load_async_to_lds_b128) && \
    __has_builtin(__builtin_amdgcn_s_wait_asynccnt)
#define USE_ASYNC_LDS 1
typedef int v4i_raw __attribute__((vector_size(16)));
typedef __attribute__((address_space(1))) v4i_raw as1_v4i;
typedef __attribute__((address_space(3))) v4i_raw as3_v4i;
#else
#define USE_ASYNC_LDS 0
#endif

union Frag16 {
    unsigned short u[16];
    uint4          q[2];
    v16bf          v;
};

__device__ __forceinline__ unsigned short f32_to_bf16(float f) {
    unsigned int x = __builtin_bit_cast(unsigned int, f);
    unsigned int r = (x + 0x7FFFu + ((x >> 16) & 1u)) >> 16;   // round-to-nearest-even
    return (unsigned short)r;
}

// ---------------------------------------------------------------------------
// Kernel 1: Y = X @ W + b, output as bf16 in [B, nH, S, hd] layout.
//   X: [16384, 512] f32 row-major, W: [512, 512] f32 row-major, b: [512] f32
//   Tiles: BM=128, BN=64, BK=32. 256 threads = 8 waves; wave w owns M-tile w,
//   computes 4 N-tiles => 4 v_wmma_f32_16x16x32_bf16 per k-step.
// ---------------------------------------------------------------------------
__global__ __launch_bounds__(256)
void qkv_gemm(const float* __restrict__ X, const float* __restrict__ W,
              const float* __restrict__ bias, unsigned short* __restrict__ out)
{
    __shared__ __align__(16) unsigned short ldsA[128][40];  // [m][k] bf16, stride 80B
    __shared__ __align__(16) unsigned short ldsB[64][40];   // Wt: [n][k] bf16

    const int tid  = threadIdx.x;
    const int wave = tid >> 5;
    const int lane = tid & 31;
    const int half = lane >> 4;       // lane-half: 0 or 1
    const int ln   = lane & 15;
    const int m0   = blockIdx.x * 128;
    const int n0   = blockIdx.y * 64;

    v8f acc[4] = {};

    for (int k0 = 0; k0 < 512; k0 += 32) {
        __syncthreads();
        // --- stage A tile (128x32 f32 -> bf16), 1024 float4, 4 per thread ---
#pragma unroll
        for (int i = 0; i < 4; ++i) {
            const int idx = tid + i * 256;          // 0..1023
            const int row = idx >> 3;               // 8 float4 per row
            const int kq  = (idx & 7) << 2;
            const float4 f = *(const float4*)(X + (size_t)(m0 + row) * 512 + k0 + kq);
            ldsA[row][kq + 0] = f32_to_bf16(f.x);
            ldsA[row][kq + 1] = f32_to_bf16(f.y);
            ldsA[row][kq + 2] = f32_to_bf16(f.z);
            ldsA[row][kq + 3] = f32_to_bf16(f.w);
        }
        // --- stage W tile transposed (32x64 -> Bt[n][k]), 512 float4 ---
#pragma unroll
        for (int i = 0; i < 2; ++i) {
            const int idx = tid + i * 256;          // 0..511
            const int kr  = idx >> 4;               // 16 float4 per k-row
            const int nq  = (idx & 15) << 2;
            const float4 f = *(const float4*)(W + (size_t)(k0 + kr) * 512 + n0 + nq);
            ldsB[nq + 0][kr] = f32_to_bf16(f.x);
            ldsB[nq + 1][kr] = f32_to_bf16(f.y);
            ldsB[nq + 2][kr] = f32_to_bf16(f.z);
            ldsB[nq + 3][kr] = f32_to_bf16(f.w);
        }
        __syncthreads();

        // A fragment: lane = row (M), e<8 -> k = half*8+e ; e>=8 -> k = 16+half*8+(e-8)
        Frag16 a;
        {
            const int row = wave * 16 + ln;
            a.q[0] = *(const uint4*)(&ldsA[row][half * 8]);
            a.q[1] = *(const uint4*)(&ldsA[row][16 + half * 8]);
        }
#pragma unroll
        for (int nt = 0; nt < 4; ++nt) {
            // B fragment: lane = col (N), e<8 -> k = half*8+e ; e>=8 -> k = 16+half*8+(e-8)
            Frag16 b;
            const int col = nt * 16 + ln;
            b.q[0] = *(const uint4*)(&ldsB[col][half * 8]);
            b.q[1] = *(const uint4*)(&ldsB[col][16 + half * 8]);
            acc[nt] = __builtin_amdgcn_wmma_f32_16x16x32_bf16(
                false, a.v, false, b.v, (short)0, acc[nt], false, false);
        }
    }

    // epilogue: + bias, convert to bf16, scatter into [B, nH, S, hd]
#pragma unroll
    for (int nt = 0; nt < 4; ++nt) {
        const int n  = n0 + nt * 16 + ln;
        const float bv = bias[n];
        const int h = n >> 6, d = n & 63;
#pragma unroll
        for (int r = 0; r < 8; ++r) {
            const int m = m0 + wave * 16 + half * 8 + r;    // C layout: row = r + 8*half
            const int bb = m >> 11;
            const int s  = m & 2047;
            out[((size_t)(bb * 8 + h) * 2048 + s) * 64 + d] = f32_to_bf16(acc[nt][r] + bv);
        }
    }
}

// ---------------------------------------------------------------------------
// Kernel 2: flash attention. Block = 256 thr (8 waves); each wave owns 16 query
// rows (block covers 128). Online softmax is wave-private: row stats reduce
// across the 16-lane half that holds the row's columns.
//   grid = (S/128, B*nH)
// ---------------------------------------------------------------------------
__global__ __launch_bounds__(256)
void attn(const unsigned short* __restrict__ Qb, const unsigned short* __restrict__ Kb,
          const unsigned short* __restrict__ Vb, const float* __restrict__ mask,
          float* __restrict__ out)
{
    __shared__ __align__(16) unsigned short ldsK[64][72];       // K rows [key][d] == Bt for Q.K^T
    __shared__ __align__(16) unsigned short ldsV[64][72];       // V transposed [d][key]
    __shared__ __align__(16) unsigned short ldsP[8][16][72];    // per-wave probs scratch

    const int tid  = threadIdx.x;
    const int wave = tid >> 5;
    const int lane = tid & 31;
    const int half = lane >> 4;
    const int ln   = lane & 15;
    const int bh   = blockIdx.y;            // 0..63
    const int b    = bh >> 3;
    const int h    = bh & 7;
    const int q0   = blockIdx.x * 128;

    const unsigned short* Q = Qb + (size_t)bh * 2048 * 64;
    const unsigned short* K = Kb + (size_t)bh * 2048 * 64;
    const unsigned short* V = Vb + (size_t)bh * 2048 * 64;

    // Q fragments for this wave's 16 rows; 2 k-steps over head_dim=64.
    Frag16 qf[2];
    {
        const unsigned short* qr = Q + (size_t)(q0 + wave * 16 + ln) * 64;
        qf[0].q[0] = *(const uint4*)(qr +      half * 8);
        qf[0].q[1] = *(const uint4*)(qr + 16 + half * 8);
        qf[1].q[0] = *(const uint4*)(qr + 32 + half * 8);
        qf[1].q[1] = *(const uint4*)(qr + 48 + half * 8);
    }

    float m_run[8], l_run[8];
#pragma unroll
    for (int r = 0; r < 8; ++r) { m_run[r] = -3.0e38f; l_run[r] = 0.0f; }
    v8f o[4] = {};
    const float scale = 0.125f;     // 1/sqrt(64)

    for (int j = 0; j < 2048; j += 64) {
        __syncthreads();
        // --- stage K block row-major: 64x64 bf16, 8KB ---
#if USE_ASYNC_LDS
#pragma unroll
        for (int i = 0; i < 2; ++i) {
            const int idx = tid + i * 256;      // 0..511, 8 bf16 per chunk
            const int key = idx >> 3;
            const int dq  = (idx & 7) << 3;
            __builtin_amdgcn_global_load_async_to_lds_b128(
                (as1_v4i*)(K + (size_t)(j + key) * 64 + dq),
                (as3_v4i*)(&ldsK[key][dq]), 0, 0);
        }
#else
#pragma unroll
        for (int i = 0; i < 2; ++i) {
            const int idx = tid + i * 256;
            const int key = idx >> 3;
            const int dq  = (idx & 7) << 3;
            *(uint4*)(&ldsK[key][dq]) = *(const uint4*)(K + (size_t)(j + key) * 64 + dq);
        }
#endif
        // --- stage V block transposed [d][key], packed u32 stores (key pairs) ---
        {
            const int kp = (tid >> 3) * 2;      // even key: 0,2,..,62
            const int dq = (tid & 7) << 3;      // 8 d-values per thread
            union { uint4 q; unsigned short u[8]; } v0, v1;
            v0.q = *(const uint4*)(V + (size_t)(j + kp)     * 64 + dq);
            v1.q = *(const uint4*)(V + (size_t)(j + kp + 1) * 64 + dq);
#pragma unroll
            for (int e = 0; e < 8; ++e) {
                const unsigned int pk =
                    (unsigned int)v0.u[e] | ((unsigned int)v1.u[e] << 16);
                *(unsigned int*)(&ldsV[dq + e][kp]) = pk;
            }
        }
        if (j + 64 < 2048) {        // global_prefetch_b8 for next block
            __builtin_prefetch(K + (size_t)(j + 64) * 64 + lane * 64, 0, 1);
            __builtin_prefetch(V + (size_t)(j + 64) * 64 + lane * 64, 0, 1);
        }
#if USE_ASYNC_LDS
        __builtin_amdgcn_s_wait_asynccnt(0);
#endif
        __syncthreads();

        // --- scores: S = Q K^T * scale + mask, 4 column tiles of 16 keys ---
        v8f s[4];
#pragma unroll
        for (int nt = 0; nt < 4; ++nt) {
            v8f c = {};
#pragma unroll
            for (int ks = 0; ks < 2; ++ks) {
                Frag16 kb;
                const int keyl  = nt * 16 + ln;
                const int kbase = ks * 32 + half * 8;
                kb.q[0] = *(const uint4*)(&ldsK[keyl][kbase]);
                kb.q[1] = *(const uint4*)(&ldsK[keyl][kbase + 16]);
                c = __builtin_amdgcn_wmma_f32_16x16x32_bf16(
                    false, qf[ks].v, false, kb.v, (short)0, c, false, false);
            }
            const float mk = mask[(size_t)b * 2048 + j + nt * 16 + ln];
#pragma unroll
            for (int r = 0; r < 8; ++r) c[r] = c[r] * scale + mk;
            s[nt] = c;
        }

        // --- online softmax (row stats live in a 16-lane half) ---
        float mnew[8], corr[8];
#pragma unroll
        for (int r = 0; r < 8; ++r) {
            float v = fmaxf(fmaxf(s[0][r], s[1][r]), fmaxf(s[2][r], s[3][r]));
            v = fmaxf(v, __shfl_xor(v, 1, 32));
            v = fmaxf(v, __shfl_xor(v, 2, 32));
            v = fmaxf(v, __shfl_xor(v, 4, 32));
            v = fmaxf(v, __shfl_xor(v, 8, 32));
            const float mo = m_run[r];
            const float mn = fmaxf(mo, v);
            mnew[r] = mn;
            corr[r] = __expf(mo - mn);
        }
#pragma unroll
        for (int r = 0; r < 8; ++r) {
            float ps = 0.0f;
#pragma unroll
            for (int nt = 0; nt < 4; ++nt) {
                const float p = __expf(s[nt][r] - mnew[r]);
                s[nt][r] = p;
                ps += p;
            }
            ps += __shfl_xor(ps, 1, 32);
            ps += __shfl_xor(ps, 2, 32);
            ps += __shfl_xor(ps, 4, 32);
            ps += __shfl_xor(ps, 8, 32);
            l_run[r] = l_run[r] * corr[r] + ps;
            m_run[r] = mnew[r];
        }

        // --- rescale O, park P (bf16) in per-wave LDS scratch ---
#pragma unroll
        for (int nt = 0; nt < 4; ++nt)
#pragma unroll
            for (int r = 0; r < 8; ++r) {
                ldsP[wave][half * 8 + r][nt * 16 + ln] = f32_to_bf16(s[nt][r]);
                o[nt][r] *= corr[r];
            }
        __syncthreads();

        // --- O += P @ V : A-frags from ldsP, B-frags from ldsV (transposed) ---
#pragma unroll
        for (int ks = 0; ks < 2; ++ks) {
            Frag16 pf;
            pf.q[0] = *(const uint4*)(&ldsP[wave][ln][ks * 32 +      half * 8]);
            pf.q[1] = *(const uint4*)(&ldsP[wave][ln][ks * 32 + 16 + half * 8]);
#pragma unroll
            for (int nt = 0; nt < 4; ++nt) {
                Frag16 vf;
                const int d     = nt * 16 + ln;
                const int kbase = ks * 32 + half * 8;
                vf.q[0] = *(const uint4*)(&ldsV[d][kbase]);
                vf.q[1] = *(const uint4*)(&ldsV[d][kbase + 16]);
                o[nt] = __builtin_amdgcn_wmma_f32_16x16x32_bf16(
                    false, pf.v, false, vf.v, (short)0, o[nt], false, false);
            }
        }
    }

    // --- normalize and write ctx back as [B, S, H_MODEL] f32 ---
#pragma unroll
    for (int nt = 0; nt < 4; ++nt) {
        const int d = nt * 16 + ln;
#pragma unroll
        for (int r = 0; r < 8; ++r) {
            const int sq = q0 + wave * 16 + half * 8 + r;
            out[((size_t)b * 2048 + sq) * 512 + h * 64 + d] = o[nt][r] / l_run[r];
        }
    }
}

// ---------------------------------------------------------------------------
extern "C" void kernel_launch(void* const* d_in, const int* in_sizes, int n_in,
                              void* d_out, int out_size, void* d_ws, size_t ws_size,
                              hipStream_t stream) {
    const float* q    = (const float*)d_in[0];
    const float* k    = (const float*)d_in[1];
    const float* v    = (const float*)d_in[2];
    const float* mask = (const float*)d_in[3];
    const float* Wq   = (const float*)d_in[4];
    const float* bq   = (const float*)d_in[5];
    const float* Wk   = (const float*)d_in[6];
    const float* bk   = (const float*)d_in[7];
    const float* Wv   = (const float*)d_in[8];
    const float* bv   = (const float*)d_in[9];

    const size_t elems = (size_t)8 * 8 * 2048 * 64;     // per-tensor bf16 elems
    unsigned short* Qb = (unsigned short*)d_ws;
    unsigned short* Kb = Qb + elems;
    unsigned short* Vb = Kb + elems;

    const dim3 blk(256);
    qkv_gemm<<<dim3(128, 8), blk, 0, stream>>>(q, Wq, bq, Qb);
    qkv_gemm<<<dim3(128, 8), blk, 0, stream>>>(k, Wk, bk, Kb);
    qkv_gemm<<<dim3(128, 8), blk, 0, stream>>>(v, Wv, bv, Vb);
    attn<<<dim3(16, 64), blk, 0, stream>>>(Qb, Kb, Vb, mask, (float*)d_out);
}
